// DynamicSlidingWindowAttention_11003706212961
// MI455X (gfx1250) — matrix operand order repaired and checked by reading, not compiled, not run on hardware
//
#include <hip/hip_runtime.h>

// DynamicSlidingWindowAttention forward, flash-attention style, gfx1250.
// B=2 H=16 T=2048 D=64, per-head window sizes (causal & q-k <= w).
// bf16 WMMA (v_wmma_f32_16x16x32_bf16) for QK^T and PV, fp32 softmax/accum.
// Round 4: double-buffered LDS (prefetch next K/V tile during compute,
// single barrier per tile), immediate-offset staging addresses, per-wave
// skip of KV tiles fully outside the wave's causal window.

typedef __attribute__((ext_vector_type(16))) __bf16 v16bf;
typedef __attribute__((ext_vector_type(8)))  __bf16 v8bf;
typedef __attribute__((ext_vector_type(8)))  float  v8f;

#define BATCH  2
#define NHEAD  16
#define SEQ    2048
#define HDIM   64
#define BQ     128    // query rows per workgroup (16 per wave)
#define BK     64     // kv columns per inner tile
#define NWAVE  8
#define NTHR   256

static __device__ __forceinline__ __bf16 f2bf(float x) {
  union { float f; unsigned u; } c; c.f = x;
  unsigned short hbits = (unsigned short)(c.u >> 16);   // truncate-to-bf16
  union { unsigned short s; __bf16 b; } r; r.s = hbits;
  return r.b;
}

__global__ __launch_bounds__(NTHR) void dswa_fwd(
    const float* __restrict__ Q, const float* __restrict__ K,
    const float* __restrict__ V, const int* __restrict__ WSZ,
    float* __restrict__ O)
{
  __shared__ alignas(128) __bf16 Kt[2][BK][HDIM];     // [buf][kv][d]  16 KB
  __shared__ alignas(128) __bf16 Vt[2][HDIM][BK];     // [buf][d][kv]  16 KB
  __shared__ alignas(128) __bf16 Pt[NWAVE][16][BK];   // per-wave P    16 KB

  const int tid  = threadIdx.x;
  const int lane = tid & 31;
  const int wave = tid >> 5;
  const int qt = blockIdx.x, h = blockIdx.y, b = blockIdx.z;
  const int q0 = qt * BQ;
  const int w  = WSZ[h];

  const size_t bh = (size_t)(b * NHEAD + h) * SEQ;
  const float scale = 0.125f;            // 1/sqrt(64), folded into Q

  const int n    = lane & 15;            // B/C fragment column, A-frag row
  const int half = lane >> 4;            // 0 or 1 (16-lane half)
  const int kb   = half << 3;            // unused (0 or 8); A-frags use drow
  const int drow = half << 4;            // B-fragment K-dim group (0 or 16)
  const int qw   = q0 + wave * 16;       // this wave's first query row

  // ---- Q A-fragments: 16x64 rows -> two 16x32 bf16 fragments (scaled) ----
  v16bf aq0, aq1;
  {
    const float* qp = Q + (bh + (size_t)(qw + n)) * HDIM;
    #pragma unroll
    for (int j = 0; j < 8; ++j) {
      aq0[j]     = f2bf(qp[      drow + j] * scale);
      aq0[j + 8] = f2bf(qp[ 8 +  drow + j] * scale);
      aq1[j]     = f2bf(qp[32 +  drow + j] * scale);
      aq1[j + 8] = f2bf(qp[40 +  drow + j] * scale);
    }
  }

  v8f o[4];                                    // O accum: 16x64 fp32
  #pragma unroll
  for (int dh = 0; dh < 4; ++dh) o[dh] = (v8f){};
  float rowM[8], rowL[8];                      // rowL is a per-lane PARTIAL sum
  #pragma unroll
  for (int r = 0; r < 8; ++r) { rowM[r] = -1e30f; rowL[r] = 0.0f; }

  int kvs = q0 - w; if (kvs < 0) kvs = 0;
  kvs &= ~(BK - 1);                            // windowed heads skip tiles here
  const int kve = q0 + BQ;                     // causal upper bound (tile-aligned)

  // ---- staging addresses: uniform base + constant lane offset + imm ----
  const int laneRow = tid >> 4;                // 0..15
  const int laneCol = (tid & 15) << 2;         // 0,4,..,60
  const unsigned laneOff = (unsigned)(laneRow * HDIM + laneCol);
  const float* kBase = K + bh * HDIM + laneOff;
  const float* vBase = V + bh * HDIM + laneOff;

  float4 kr[4], vr[4];                         // prefetch registers
  auto ldRegs = [&](int kvb_) {
    const float* kp = kBase + (size_t)kvb_ * HDIM;
    const float* vp = vBase + (size_t)kvb_ * HDIM;
    #pragma unroll
    for (int i = 0; i < 4; ++i) {              // imm offsets 0/4/8/12 KB
      kr[i] = *(const float4*)(kp + i * (16 * HDIM));
      vr[i] = *(const float4*)(vp + i * (16 * HDIM));
    }
  };
  auto stLDS = [&](int buf_) {
    #pragma unroll
    for (int i = 0; i < 4; ++i) {
      const int row = laneRow + 16 * i, col = laneCol;
      Kt[buf_][row][col + 0] = f2bf(kr[i].x); Kt[buf_][row][col + 1] = f2bf(kr[i].y);
      Kt[buf_][row][col + 2] = f2bf(kr[i].z); Kt[buf_][row][col + 3] = f2bf(kr[i].w);
      Vt[buf_][col + 0][row] = f2bf(vr[i].x); Vt[buf_][col + 1][row] = f2bf(vr[i].y);
      Vt[buf_][col + 2][row] = f2bf(vr[i].z); Vt[buf_][col + 3][row] = f2bf(vr[i].w);
    }
  };

  ldRegs(kvs);                                 // prologue prefetch
  int buf = 0;

  for (int kvb = kvs; kvb < kve; kvb += BK, buf ^= 1) {
    // Write the staged tile. Safe: the last readers of this buffer ran at
    // iteration kvb-2*BK, and every wave passed the barrier of kvb-BK (which
    // follows its compute of kvb-2*BK) before arriving here.
    stLDS(buf);
    if (kvb + BK < kve) ldRegs(kvb + BK);      // prefetch next tile (in flight
                                               // during this tile's compute)
    __syncthreads();                           // buffer `buf` is ready

    // Per-wave relevance: this wave's KV union is [qw - w, qw + 15].
    // Tiles fully outside contribute exactly zero -> skip compute.
    if ((kvb > qw + 15) || (kvb + (BK - 1) < qw - w)) continue;

    // ---- S = Q K^T : four 16x16 fp32 C tiles (kv quarters), K-dim = d ----
    v8f s[4];
    #pragma unroll
    for (int nh = 0; nh < 4; ++nh) {
      const v16bf bk0 = *(const v16bf*)&Kt[buf][nh * 16 + n][ 0 + drow];
      const v16bf bk1 = *(const v16bf*)&Kt[buf][nh * 16 + n][32 + drow];
      v8f acc = (v8f){};
      acc = __builtin_amdgcn_wmma_f32_16x16x32_bf16(false, aq0, false, bk0, (short)0, acc, false, false);
      acc = __builtin_amdgcn_wmma_f32_16x16x32_bf16(false, aq1, false, bk1, (short)0, acc, false, false);
      s[nh] = acc;
    }

    // Wave-uniform: tile fully inside the causal window for all 16 rows?
    const bool interior = ((kvb + (BK - 1)) <= qw) && ((qw + 15 - kvb) <= w);

    // ---- mask (edge tiles only) ----
    if (!interior) {
      #pragma unroll
      for (int nh = 0; nh < 4; ++nh) {
        const int kvg = kvb + nh * 16 + n;
        #pragma unroll
        for (int r = 0; r < 8; ++r) {
          const int qg = qw + r + (half << 3);
          if (!((qg >= kvg) && (qg - kvg <= w))) s[nh][r] = -1e30f;
        }
      }
    }

    // ---- running max (cross-lane reduce within 16-lane half) ----
    float mnew[8];
    #pragma unroll
    for (int r = 0; r < 8; ++r) {
      float m = rowM[r];
      m = fmaxf(m, fmaxf(fmaxf(s[0][r], s[1][r]), fmaxf(s[2][r], s[3][r])));
      #pragma unroll
      for (int off = 1; off < 16; off <<= 1)
        m = fmaxf(m, __shfl_xor(m, off, 32));
      mnew[r] = m;
    }

    float corr[8];
    #pragma unroll
    for (int r = 0; r < 8; ++r) {
      corr[r] = __expf(rowM[r] - mnew[r]);
      rowM[r] = mnew[r];
    }

    // ---- P = exp(S - mnew); lane-private partial row sums (no reduce) ----
    if (interior) {
      #pragma unroll
      for (int r = 0; r < 8; ++r) {
        const int mrow = r + (half << 3);
        float psum = 0.0f;
        #pragma unroll
        for (int nh = 0; nh < 4; ++nh) {
          const float p = __expf(s[nh][r] - mnew[r]);
          Pt[wave][mrow][nh * 16 + n] = f2bf(p);
          psum += p;
        }
        rowL[r] = rowL[r] * corr[r] + psum;
        #pragma unroll
        for (int dh = 0; dh < 4; ++dh) o[dh][r] *= corr[r];
      }
    } else {
      #pragma unroll
      for (int r = 0; r < 8; ++r) {
        const int mrow = r + (half << 3);
        float psum = 0.0f;
        #pragma unroll
        for (int nh = 0; nh < 4; ++nh) {
          const float sv = s[nh][r];
          const float p = (sv > -1e29f) ? __expf(sv - mnew[r]) : 0.0f;
          Pt[wave][mrow][nh * 16 + n] = f2bf(p);
          psum += p;
        }
        rowL[r] = rowL[r] * corr[r] + psum;
        #pragma unroll
        for (int dh = 0; dh < 4; ++dh) o[dh][r] *= corr[r];
      }
    }

    // ---- reload P as A-fragments (C-layout -> A-layout via per-wave LDS) ----
    // LDS ops are in-order within a wave: RAW across lanes of this wave is safe.
    v16bf ap0, ap1;
    {
      const v8bf l0 = *(const v8bf*)&Pt[wave][n][     drow];
      const v8bf h0 = *(const v8bf*)&Pt[wave][n][ 8 + drow];
      const v8bf l1 = *(const v8bf*)&Pt[wave][n][32 + drow];
      const v8bf h1 = *(const v8bf*)&Pt[wave][n][40 + drow];
      #pragma unroll
      for (int j = 0; j < 8; ++j) {
        ap0[j] = l0[j]; ap0[j + 8] = h0[j];
        ap1[j] = l1[j]; ap1[j + 8] = h1[j];
      }
    }

    // ---- O += P V : K-dim = kv (64), B-frags from transposed Vt ----
    #pragma unroll
    for (int dh = 0; dh < 4; ++dh) {
      const v16bf bv0 = *(const v16bf*)&Vt[buf][dh * 16 + n][ 0 + drow];
      const v16bf bv1 = *(const v16bf*)&Vt[buf][dh * 16 + n][32 + drow];
      o[dh] = __builtin_amdgcn_wmma_f32_16x16x32_bf16(false, ap0, false, bv0, (short)0, o[dh], false, false);
      o[dh] = __builtin_amdgcn_wmma_f32_16x16x32_bf16(false, ap1, false, bv1, (short)0, o[dh], false, false);
    }
  }

  // ---- final: reduce partial row sums once, normalize, store fp32 ----
  #pragma unroll
  for (int r = 0; r < 8; ++r) {
    float ls = rowL[r];
    #pragma unroll
    for (int off = 1; off < 16; off <<= 1)
      ls += __shfl_xor(ls, off, 32);
    const float inv = 1.0f / ls;
    const int mrow = r + (half << 3);
    float* op = O + (bh + (size_t)(qw + mrow)) * HDIM;
    op[ 0 + n] = o[0][r] * inv;
    op[16 + n] = o[1][r] * inv;
    op[32 + n] = o[2][r] * inv;
    op[48 + n] = o[3][r] * inv;
  }
}

extern "C" void kernel_launch(void* const* d_in, const int* in_sizes, int n_in,
                              void* d_out, int out_size, void* d_ws, size_t ws_size,
                              hipStream_t stream) {
  (void)in_sizes; (void)n_in; (void)out_size; (void)d_ws; (void)ws_size;
  const float* Q  = (const float*)d_in[0];
  const float* K  = (const float*)d_in[1];
  const float* V  = (const float*)d_in[2];
  const int*   WS = (const int*)d_in[3];
  float* O = (float*)d_out;
  dim3 grid(SEQ / BQ, NHEAD, BATCH);
  dim3 block(NTHR);
  hipLaunchKernelGGL(dswa_fwd, grid, block, 0, stream, Q, K, V, WS, O);
}
